// Node_only_Type_75900662055233
// MI455X (gfx1250) — matrix operand order repaired and checked by reading, not compiled, not run on hardware
//
#include <hip/hip_runtime.h>
#include <stdint.h>

#define N_NODES 50000
#define N_EDGES 800000
#define IN_DIM  100
#define MTILES  3125   // 50000 / 16
#define NPW     4      // 16x16 N-tiles per wave (wave computes 16x64)

typedef __attribute__((ext_vector_type(16))) __bf16 v16bf;
typedef __attribute__((ext_vector_type(8)))  float  v8f;

union Frag16 { uint4 q[2]; v16bf v; };

__device__ __forceinline__ unsigned short f32_to_bf16(float f) {
  union { float f; unsigned int u; } c; c.f = f;
  unsigned int u = c.u;
  unsigned int r = (u + 0x7FFFu + ((u >> 16) & 1u)) >> 16;  // round-nearest-even
  return (unsigned short)r;
}

// ---------------- graph prep ----------------
__global__ void k_degree(const int* __restrict__ ei, float* __restrict__ deg) {
  int e = blockIdx.x * 256 + threadIdx.x;
  if (e < N_EDGES) atomicAdd(&deg[ei[N_EDGES + e]], 1.0f);
}

__global__ void k_dinv(const float* __restrict__ deg, float* __restrict__ dinv) {
  int i = blockIdx.x * 256 + threadIdx.x;
  if (i < N_NODES) dinv[i] = rsqrtf(deg[i] + 1.0f);   // +1 self-loop; always > 0
}

__global__ void k_norm(const int* __restrict__ ei, const float* __restrict__ dinv,
                       float* __restrict__ nrm) {
  int e = blockIdx.x * 256 + threadIdx.x;
  if (e < N_EDGES) nrm[e] = dinv[ei[e]] * dinv[ei[N_EDGES + e]];
}

// ---------------- input conversion (fp32 -> padded bf16) ----------------
__global__ void k_convert_x(const float* __restrict__ x, unsigned short* __restrict__ A) {
  unsigned int idx = blockIdx.x * 256u + threadIdx.x;
  if (idx >= (unsigned)N_NODES * 128u) return;
  unsigned int i = idx >> 7, k = idx & 127u;
  float v = (k < IN_DIM) ? x[i * IN_DIM + k] : 0.0f;
  A[idx] = f32_to_bf16(v);
}

// ---------------- weight pack into WMMA B-fragment layout ----------------
// packed[(nt*KT + kt)*512 + lane*16 + j] = W[kt*32 + ((j>>3)<<4) + ((lane>>4)<<3) + (j&7)][nt*16 + (lane&15)]
__global__ void k_pack_w(const float* __restrict__ W, unsigned short* __restrict__ Bp,
                         int K, int Nn, int Kpad, int Npad) {
  int KT = Kpad >> 5, NT = Npad >> 4;
  int total = KT * NT * 512;
  int idx = blockIdx.x * 256 + threadIdx.x;
  if (idx >= total) return;
  int j    = idx & 15;
  int lane = (idx >> 4) & 31;
  int t    = idx >> 9;
  int kt   = t % KT;
  int nt   = t / KT;
  int n = nt * 16 + (lane & 15);
  int k = kt * 32 + ((j >> 3) << 4) + ((lane >> 4) << 3) + (j & 7);
  float v = (k < K && n < Nn) ? W[(size_t)k * Nn + n] : 0.0f;
  Bp[idx] = f32_to_bf16(v);
}

// ---------------- bf16 WMMA GEMM: H = A(bf16,[50000 x Kpad]) @ Wp -> fp32 [50000 x OutPad]
// one wave computes a 16x64 output slab (4 accumulators reuse one A fragment);
// 8 waves / block along M => block covers 128 x 64
__global__ void k_gemm_wmma(const unsigned short* __restrict__ A,
                            const unsigned short* __restrict__ Bp,
                            float* __restrict__ C, int Kpad, int OutPad) {
  int lane = threadIdx.x & 31;
  int wave = threadIdx.x >> 5;
  int mt  = blockIdx.x * 8 + wave;
  int ntb = blockIdx.y * NPW;
  if (mt >= MTILES) return;                      // wave-uniform exit: EXEC stays all-1s
  int KT = Kpad >> 5;

  int row = mt * 16 + (lane & 15);
  int kb  = (lane >> 4) << 3;                    // 0 or 8
  const unsigned short* Arow = A + (size_t)row * Kpad + kb;
  const unsigned short* B0   = Bp + (size_t)ntb * KT * 512 + (size_t)lane * 16;
  const size_t tileStride = (size_t)KT * 512;    // stride between adjacent N-tiles

  v8f acc0 = {}, acc1 = {}, acc2 = {}, acc3 = {};
  for (int kt = 0; kt < KT; ++kt) {
    Frag16 a;
    a.q[0] = *(const uint4*)(Arow + kt * 32);        // K = kb..kb+7
    a.q[1] = *(const uint4*)(Arow + kt * 32 + 16);   // K = kb+16..kb+23
    const unsigned short* bt = B0 + (size_t)kt * 512;
    Frag16 b;
    b.q[0] = *(const uint4*)(bt);
    b.q[1] = *(const uint4*)(bt + 8);
    acc0 = __builtin_amdgcn_wmma_f32_16x16x32_bf16(false, a.v, false, b.v, (short)0, acc0, false, false);
    bt += tileStride;
    b.q[0] = *(const uint4*)(bt);
    b.q[1] = *(const uint4*)(bt + 8);
    acc1 = __builtin_amdgcn_wmma_f32_16x16x32_bf16(false, a.v, false, b.v, (short)0, acc1, false, false);
    bt += tileStride;
    b.q[0] = *(const uint4*)(bt);
    b.q[1] = *(const uint4*)(bt + 8);
    acc2 = __builtin_amdgcn_wmma_f32_16x16x32_bf16(false, a.v, false, b.v, (short)0, acc2, false, false);
    bt += tileStride;
    b.q[0] = *(const uint4*)(bt);
    b.q[1] = *(const uint4*)(bt + 8);
    acc3 = __builtin_amdgcn_wmma_f32_16x16x32_bf16(false, a.v, false, b.v, (short)0, acc3, false, false);
  }

  int rbase = mt * 16 + ((lane >> 4) << 3);
  int col0  = ntb * 16 + (lane & 15);
  float* Cc = C + (size_t)rbase * OutPad + col0;
#pragma unroll
  for (int r = 0; r < 8; ++r) {
    float* cr = Cc + (size_t)r * OutPad;
    cr[0]  = acc0[r];
    cr[16] = acc1[r];
    cr[32] = acc2[r];
    cr[48] = acc3[r];
  }
}

// ---------------- edge scatter: agg[dst] += h[src] * norm ----------------
__global__ void k_scatter(const int* __restrict__ ei, const float* __restrict__ nrm,
                          const float* __restrict__ h, float* __restrict__ agg,
                          int Out, int OutPad) {
  unsigned int idx = blockIdx.x * 256u + threadIdx.x;
  unsigned int total = (unsigned)N_EDGES * (unsigned)Out;
  if (idx >= total) return;
  unsigned int e = idx / (unsigned)Out;
  unsigned int f = idx - e * (unsigned)Out;
  int s = ei[e], d = ei[N_EDGES + e];
  atomicAdd(&agg[(size_t)d * OutPad + f], h[(size_t)s * OutPad + f] * nrm[e]);
}

// ---------------- finalize: out = relu(agg + h*dinv^2 + b); bf16 and/or fp32 out
__global__ void k_finalize(const float* __restrict__ agg, const float* __restrict__ h,
                           const float* __restrict__ dinv, const float* __restrict__ bias,
                           unsigned short* __restrict__ nextA, float* __restrict__ outF,
                           int Out, int OutPad) {
  unsigned int idx = blockIdx.x * 256u + threadIdx.x;
  if (idx >= (unsigned)N_NODES * (unsigned)OutPad) return;
  unsigned int i = idx / (unsigned)OutPad;
  unsigned int f = idx - i * (unsigned)OutPad;
  float v = 0.0f;
  if (f < (unsigned)Out) {
    float di = dinv[i];
    v = fmaxf(agg[idx] + h[idx] * di * di + bias[f], 0.0f);
  }
  if (nextA) nextA[idx] = f32_to_bf16(v);
  if (outF)  outF[idx]  = v;
}

// ---------------- group classification (prototype match, reference precedence)
__global__ void k_group(const float* __restrict__ x, const float* __restrict__ nf,
                        int* __restrict__ grp) {
  int i = blockIdx.x * 256 + threadIdx.x;
  if (i >= N_NODES) return;
  bool e0 = true, e1 = true, e2 = true;
  const float* xr = x + (size_t)i * IN_DIM;
  for (int k = 0; k < IN_DIM; ++k) {
    float xv = xr[k];
    e0 &= (xv == nf[k]);
    e1 &= (xv == nf[IN_DIM + k]);
    e2 &= (xv == nf[2 * IN_DIM + k]);
  }
  grp[i] = e0 ? 0 : (e1 ? 3 : (e2 ? 1 : 2));
}

// ---------------- per-block LDS partial sums, then global atomic reduce
#define MEAN_NODES 128
__global__ void k_group_sums(const float* __restrict__ out4, const int* __restrict__ grp,
                             float* __restrict__ sums, float* __restrict__ cnts,
                             int Out, int OutPad) {
  __shared__ float ls[4 * 200];
  __shared__ float lc[4];
  int tid = threadIdx.x;
  for (int j = tid; j < 4 * 200; j += 256) ls[j] = 0.0f;
  if (tid < 4) lc[tid] = 0.0f;
  __syncthreads();
  int n0 = blockIdx.x * MEAN_NODES;
  for (int t = 0; t < MEAN_NODES; ++t) {
    int n = n0 + t;
    if (n >= N_NODES) break;
    int g = grp[n];
    if (tid < Out) ls[g * Out + tid] += out4[(size_t)n * OutPad + tid];
    if (tid == 0)  lc[g] += 1.0f;
  }
  __syncthreads();
  for (int j = tid; j < 4 * Out; j += 256) atomicAdd(&sums[j], ls[j]);
  if (tid < 4) atomicAdd(&cnts[tid], lc[tid]);
}

__global__ void k_means(const float* __restrict__ sums, const float* __restrict__ cnts,
                        float* __restrict__ out, int Out) {
  int j = blockIdx.x * 256 + threadIdx.x;
  if (j >= 4 * Out) return;
  float c = cnts[j / Out];
  out[j] = (c > 0.0f) ? sums[j] / fmaxf(c, 1.0f) : 0.0f;
}

// ================= host orchestration =================
extern "C" void kernel_launch(void* const* d_in, const int* in_sizes, int n_in,
                              void* d_out, int out_size, void* d_ws, size_t ws_size,
                              hipStream_t stream) {
  const float* x  = (const float*)d_in[0];
  const float* nf = (const float*)d_in[1];
  const int*   ei = (const int*)d_in[2];
  const float* W[4] = {(const float*)d_in[3], (const float*)d_in[5],
                       (const float*)d_in[7], (const float*)d_in[9]};
  const float* B[4] = {(const float*)d_in[4], (const float*)d_in[6],
                       (const float*)d_in[8], (const float*)d_in[10]};

  // layer dims: Kin, Kpad, Out, OutPad (OutPad multiple of 64 for NPW=4 blocking)
  const int Kpad[4]   = {128, 512, 256, 128};
  const int Kin[4]    = {100, 512, 256, 128};
  const int Out[4]    = {512, 256, 128, 200};
  const int OutPad[4] = {512, 256, 128, 256};

  char* p = (char*)d_ws;
  auto alloc = [&](size_t bytes) -> void* {
    void* r = (void*)p; p += (bytes + 255) & ~(size_t)255; return r;
  };
  unsigned short* A0 = (unsigned short*)alloc((size_t)N_NODES * 512 * 2);
  unsigned short* A1 = (unsigned short*)alloc((size_t)N_NODES * 512 * 2);
  float* H   = (float*)alloc((size_t)N_NODES * 512 * 4);
  float* G   = (float*)alloc((size_t)N_NODES * 512 * 4);
  unsigned short* Wp[4];
  for (int l = 0; l < 4; ++l)
    Wp[l] = (unsigned short*)alloc((size_t)(Kpad[l] >> 5) * (OutPad[l] >> 4) * 512 * 2);
  float* deg  = (float*)alloc((size_t)N_NODES * 4);
  float* dinv = (float*)alloc((size_t)N_NODES * 4);
  float* nrm  = (float*)alloc((size_t)N_EDGES * 4);
  int*   grp  = (int*)alloc((size_t)N_NODES * 4);
  float* sums = (float*)alloc(4 * 200 * 4);
  float* cnts = (float*)alloc(4 * 4);

  // --- graph normalization ---
  hipMemsetAsync(deg, 0, (size_t)N_NODES * 4, stream);
  k_degree<<<(N_EDGES + 255) / 256, 256, 0, stream>>>(ei, deg);
  k_dinv<<<(N_NODES + 255) / 256, 256, 0, stream>>>(deg, dinv);
  k_norm<<<(N_EDGES + 255) / 256, 256, 0, stream>>>(ei, dinv, nrm);

  // --- pack weights, convert input, classify groups ---
  for (int l = 0; l < 4; ++l) {
    int total = (Kpad[l] >> 5) * (OutPad[l] >> 4) * 512;
    k_pack_w<<<(total + 255) / 256, 256, 0, stream>>>(W[l], Wp[l], Kin[l], Out[l],
                                                      Kpad[l], OutPad[l]);
  }
  {
    unsigned int total = (unsigned)N_NODES * 128u;
    k_convert_x<<<(total + 255) / 256, 256, 0, stream>>>(x, A0);
  }
  k_group<<<(N_NODES + 255) / 256, 256, 0, stream>>>(x, nf, grp);

  // --- 4 GCN layers ---
  unsigned short* Ain  = A0;
  unsigned short* Anxt = A1;
  for (int l = 0; l < 4; ++l) {
    dim3 gg((MTILES + 7) / 8, OutPad[l] >> 6);   // grid.y = OutPad / (16*NPW)
    k_gemm_wmma<<<gg, 256, 0, stream>>>(Ain, Wp[l], H, Kpad[l], OutPad[l]);

    hipMemsetAsync(G, 0, (size_t)N_NODES * OutPad[l] * 4, stream);
    unsigned int stotal = (unsigned)N_EDGES * (unsigned)Out[l];
    k_scatter<<<(stotal + 255u) / 256u, 256, 0, stream>>>(ei, nrm, H, G, Out[l], OutPad[l]);

    unsigned int ftotal = (unsigned)N_NODES * (unsigned)OutPad[l];
    if (l < 3) {
      k_finalize<<<(ftotal + 255u) / 256u, 256, 0, stream>>>(
          G, H, dinv, B[l], Anxt, (float*)nullptr, Out[l], OutPad[l]);
      unsigned short* t = Ain; Ain = Anxt; Anxt = t;
    } else {
      // final layer: fp32 result written in-place over the agg buffer
      k_finalize<<<(ftotal + 255u) / 256u, 256, 0, stream>>>(
          G, H, dinv, B[l], (unsigned short*)nullptr, G, Out[l], OutPad[l]);
    }
  }

  // --- grouped means -> d_out [1, 800] ---
  hipMemsetAsync(sums, 0, 4 * 200 * 4, stream);
  hipMemsetAsync(cnts, 0, 4 * 4, stream);
  k_group_sums<<<(N_NODES + MEAN_NODES - 1) / MEAN_NODES, 256, 0, stream>>>(
      G, grp, sums, cnts, 200, 256);
  k_means<<<(800 + 255) / 256, 256, 0, stream>>>(sums, cnts, (float*)d_out, 200);
}
